// GraphClassifier_60172491817782
// MI455X (gfx1250) — matrix operand order repaired and checked by reading, not compiled, not run on hardware
//
#include <hip/hip_runtime.h>
#include <hip/hip_bf16.h>

// ---------------------------------------------------------------------------
// GatedGraphConv graph classifier for MI455X (gfx1250), wave32.
// - All big matmuls via v_wmma_f32_16x16x32_bf16 (f32 accumulate).
// - Weights kept n-major bf16 in global so each k-step's B tile is a
//   rectangular TDM tile; tensor_load_to_lds DMAs it into LDS with hardware
//   padding (interval 16 dwords + 1 dword) == conflict-free stride 34 u16.
// - scatter-add moved BEFORE the conv GEMM (linearity of segment_sum).
// Workspace: 3 x 256MB node buffers + ~16MB bf16 weights + tails (~822MB).
// ---------------------------------------------------------------------------

#define N_NODES  65536
#define N_EDGES  131072
#define N_GRAPHS 256
#define OUT_CH   1024
#define HID      512

typedef __attribute__((ext_vector_type(16))) __bf16 bf16x16;
typedef __attribute__((ext_vector_type(8)))  float  f32x8;
typedef __attribute__((ext_vector_type(4)))  unsigned int u32x4;
typedef __attribute__((ext_vector_type(8)))  int  i32x8;
typedef __attribute__((ext_vector_type(4)))  int  i32x4;

union FragAB { bf16x16 v; unsigned int u[8]; };
union FragC  { f32x8 v;   float f[8]; };

__device__ __forceinline__ unsigned short f32_to_bf16(float f) {
    union { float f; unsigned int u; } v; v.f = f;
    unsigned int u = v.u;
    unsigned int round = ((u >> 16) & 1u) + 0x7FFFu;   // round-to-nearest-even
    return (unsigned short)((u + round) >> 16);
}
__device__ __forceinline__ unsigned int pack_bf16x2(float a, float b) {
    return (unsigned int)f32_to_bf16(a) | ((unsigned int)f32_to_bf16(b) << 16);
}
__device__ __forceinline__ float sigmoidf(float x) {
    return 1.0f / (1.0f + __expf(-x));
}

#define LDSP 34   // LDS row stride in u16 (17 dwords, bank-conflict free)

// ---------------------------------------------------------------------------
// Tensor Data Mover: load `slices` x `rows` x 32 bf16 tile (row stride K
// elements, slice stride slice_stride elements) into LDS at lds_byte_off,
// padded 1 dword per 16 dwords -> LDS row stride = 34 u16.
// 6-arg builtin variant (clang-23 / therock-10.0): extra int32x8 arg (unused
// descriptor slot, zero-filled), then cpol.
// ---------------------------------------------------------------------------
#if __has_builtin(__builtin_amdgcn_tensor_load_to_lds)
#define HAVE_TDM 1
__device__ __forceinline__ void tdm_load_bf16_tile(
        unsigned lds_byte_off, const unsigned short* gptr,
        int rows, int slices, unsigned long long slice_stride, int K) {
    unsigned long long ga = (unsigned long long)(uintptr_t)gptr;
    u32x4 g0 = { 1u,                                 // count=1 (valid D#)
                 lds_byte_off,                       // lds_addr (bytes)
                 (unsigned)(ga & 0xffffffffull),     // global_addr[31:0]
                 (unsigned)((ga >> 32) & 0x1ffffffull) | (2u << 30) }; // +type=2
    int d0 = (1 << 16)      // data_size = 2 bytes
           | (1 << 20)      // pad_enable
           | (3 << 22);     // pad_interval: 16 dwords
                            // pad_amount = 0 -> 1 dword
    int d1 = (K & 0xffff) << 16;                          // tensor_dim0[15:0]
    int d2 = ((K >> 16) & 0xffff) | ((rows & 0xffff) << 16); // dim0 hi | dim1 lo
    int d3 = ((rows >> 16) & 0xffff) | (32 << 16);        // dim1 hi | tile_dim0
    int d4 = (rows & 0xffff) | (slices << 16);            // tile_dim1 | tile_dim2
    int d5 = K;                                           // tensor_dim0_stride lo
    int d6 = (int)((slice_stride & 0xffffull) << 16);     // dim0str hi | dim1str lo
    int d7 = (int)((slice_stride >> 16) & 0xffffffffull); // tensor_dim1_stride hi
    i32x8 g1 = { d0, d1, d2, d3, d4, d5, d6, d7 };
    i32x4 g2 = { slices, 0, 0, 0 };                       // tensor_dim2
    i32x4 g3 = { 0, 0, 0, 0 };
    i32x8 g4 = { 0, 0, 0, 0, 0, 0, 0, 0 };                // unused slot
    __builtin_amdgcn_tensor_load_to_lds(g0, g1, g2, g3, g4, 0);
}
#else
#define HAVE_TDM 0
#endif

// ---------------------------------------------------------------------------
// 1) h init: h[:, :32] = embed[id], h[:, 32:64] = x[:, 1:], rest zero.
// ---------------------------------------------------------------------------
__global__ void init_h_kernel(const float* __restrict__ x,
                              const float* __restrict__ embed,
                              float* __restrict__ h) {
    int node = blockIdx.x;
    const float* xr = x + (size_t)node * 33;
    int id = (int)xr[0];
    id = id < 0 ? 0 : (id > 127 ? 127 : id);
    float* hr = h + (size_t)node * OUT_CH;
    for (int c = threadIdx.x; c < OUT_CH; c += blockDim.x) {
        float v = 0.0f;
        if (c < 32)       v = embed[id * 32 + c];
        else if (c < 64)  v = xr[1 + (c - 32)];
        hr[c] = v;
    }
}

// ---------------------------------------------------------------------------
// 2) weight conversion to bf16.
//    conv_w: transpose to n-major cwT[l][n][k] = conv_w[l][k][n].
//    w_ih / w_hh ([3072,1024] row-major) are ALREADY n-major for x @ W.T.
// ---------------------------------------------------------------------------
__global__ void to_bf16_kernel(const float* __restrict__ in,
                               unsigned short* __restrict__ out, int n) {
    int i = blockIdx.x * blockDim.x + threadIdx.x;
    if (i < n) out[i] = f32_to_bf16(in[i]);
}

__global__ void conv_w_transpose_bf16_kernel(const float* __restrict__ in,
                                             unsigned short* __restrict__ out) {
    int i = blockIdx.x * blockDim.x + threadIdx.x;   // out index (coalesced)
    int l = i >> 20;
    int r = i & 0xFFFFF;
    int n = r >> 10;
    int k = r & 1023;
    out[i] = f32_to_bf16(in[((size_t)l << 20) + ((size_t)k << 10) + n]);
}

// ---------------------------------------------------------------------------
// 3) edge scatter-add:  out[dst[e]] += h[src[e]]   (global_atomic_add_f32)
// ---------------------------------------------------------------------------
__global__ void scatter_add_kernel(const float* __restrict__ h,
                                   const int* __restrict__ src,
                                   const int* __restrict__ dst,
                                   float* __restrict__ out) {
    int e = blockIdx.x;
    int s = src[e], d = dst[e];
    const float* hp = h + (size_t)s * OUT_CH;
    float* op = out + (size_t)d * OUT_CH;
    for (int c = threadIdx.x; c < OUT_CH; c += blockDim.x)
        unsafeAtomicAdd(&op[c], hp[c]);
}

// ---------------------------------------------------------------------------
// 4) WMMA GEMM: C[M,N] = bf16(A[M,K]) @ Bt, Bt n-major bf16 [N][K].
//    Block 64x128, 8 waves; wave = 16 rows x 64 cols (4 WMMA n-tiles).
//    B tile loaded by TDM (wave 0) while all threads convert/stage A.
// ---------------------------------------------------------------------------
#define GEMM_BM 64
#define GEMM_BN 128

__global__ __launch_bounds__(256)
void gemm_bf16_wmma(const float* __restrict__ A,
                    const unsigned short* __restrict__ Bt,   // [N][K] bf16
                    float* __restrict__ C,
                    int M, int N, int K) {
    __shared__ unsigned short Asm[GEMM_BM * LDSP];
    __shared__ unsigned short Bsm[GEMM_BN * LDSP];

    int tid  = threadIdx.x;
    int wave = tid >> 5;
    int lane = tid & 31;
    int ln = lane & 15, lg = lane >> 4;

    int m0 = blockIdx.y * GEMM_BM;
    int n0 = blockIdx.x * GEMM_BN;
    int wm = (wave >> 1) * 16;   // wave row offset: 0,16,32,48
    int wn = (wave & 1) * 64;    // wave col offset: 0 or 64

#if HAVE_TDM
    unsigned bsm_off = (unsigned)(uintptr_t)(void*)&Bsm[0];
#endif

    FragC acc[4];
    for (int t = 0; t < 4; t++)
        for (int r = 0; r < 8; r++) acc[t].f[r] = 0.0f;

    for (int k0 = 0; k0 < K; k0 += 32) {
        __syncthreads();
#if HAVE_TDM
        if (wave == 0)   // async DMA of B tile: 128 rows x 32 bf16, stride K
            tdm_load_bf16_tile(bsm_off, Bt + (size_t)n0 * K + k0,
                               GEMM_BN, 1, 0ull, K);
#else
        for (int j = 0; j < 2; j++) {
            int chunk = tid + j * 256;
            int row = chunk >> 2, q = chunk & 3;
            const unsigned short* sp = Bt + (size_t)(n0 + row) * K + k0 + q * 8;
            uint4 v = *(const uint4*)sp;
            unsigned int* d = (unsigned int*)&Bsm[row * LDSP + q * 8];
            d[0] = v.x; d[1] = v.y; d[2] = v.z; d[3] = v.w;
        }
#endif
        // stage A (f32 -> bf16): 64 rows x 32 k, overlapped with the DMA
        {
            int r = tid >> 2;            // 0..63
            int q = (tid & 3) * 8;       // k sub-block
            const float* sp = A + (size_t)(m0 + r) * K + k0 + q;
            float4 f0 = *(const float4*)(sp);
            float4 f1 = *(const float4*)(sp + 4);
            unsigned int* d = (unsigned int*)&Asm[r * LDSP + q];
            d[0] = pack_bf16x2(f0.x, f0.y);
            d[1] = pack_bf16x2(f0.z, f0.w);
            d[2] = pack_bf16x2(f1.x, f1.y);
            d[3] = pack_bf16x2(f1.z, f1.w);
        }
#if HAVE_TDM
        if (wave == 0) __builtin_amdgcn_s_wait_tensorcnt(0);
#endif
        __syncthreads();

        // A fragment (16x32): lane = M, VGPR i <-> K pair (ISA 7.12.2)
        FragAB a;
        {
            const unsigned short* row = &Asm[(wm + ln) * LDSP];
            #pragma unroll
            for (int i = 0; i < 8; i++) {
                int kb = 2 * i + 16 * lg;
                a.u[i] = *(const unsigned int*)&row[kb];
            }
        }
        #pragma unroll
        for (int nt = 0; nt < 4; nt++) {
            FragAB b;
            const unsigned short* row = &Bsm[(wn + nt * 16 + ln) * LDSP];
            #pragma unroll
            for (int i = 0; i < 8; i++)
                b.u[i] = *(const unsigned int*)&row[2 * i + 16 * lg];
            acc[nt].v = __builtin_amdgcn_wmma_f32_16x16x32_bf16(
                false, a.v, false, b.v, (short)0, acc[nt].v, false, false);
        }
    }

    #pragma unroll
    for (int nt = 0; nt < 4; nt++) {
        int c = n0 + wn + nt * 16 + ln;
        #pragma unroll
        for (int r = 0; r < 8; r++) {
            int m = m0 + wm + r + 8 * lg;
            C[(size_t)m * N + c] = acc[nt].f[r];
        }
    }
}

// ---------------------------------------------------------------------------
// 5) Fused GRU cell: 32-row x 64-channel block, 6 WMMA accumulators
//    (i_r,i_z,i_n from agg@w_ih.T; h_r,h_z,h_n from h@w_hh.T), gates in
//    epilogue. gi/gh ([N,3072]) never materialized. Weight tiles arrive via
//    two 3-slice TDM loads (slice = gate, slice stride 1024*1024 elements).
// ---------------------------------------------------------------------------
#define GRU_BM 32
#define GRU_BC 64

__global__ __launch_bounds__(256)
void gru_fused_wmma(const float* __restrict__ agg,
                    const float* __restrict__ h,
                    const unsigned short* __restrict__ wih16,   // [3072][1024]
                    const unsigned short* __restrict__ whh16,   // [3072][1024]
                    const float* __restrict__ b_ih,
                    const float* __restrict__ b_hh,
                    float* __restrict__ hout) {
    __shared__ unsigned short Aag[GRU_BM * LDSP];
    __shared__ unsigned short Ahh[GRU_BM * LDSP];
    __shared__ unsigned short Bwi[3 * GRU_BC * LDSP];
    __shared__ unsigned short Bwh[3 * GRU_BC * LDSP];

    int tid  = threadIdx.x;
    int wave = tid >> 5;
    int lane = tid & 31;
    int ln = lane & 15, lg = lane >> 4;

    int m0 = blockIdx.y * GRU_BM;
    int c0 = blockIdx.x * GRU_BC;
    int wm = (wave >> 2) * 16;   // 0 or 16
    int wc = (wave & 3) * 16;    // 0,16,32,48

#if HAVE_TDM
    unsigned bwi_off = (unsigned)(uintptr_t)(void*)&Bwi[0];
    unsigned bwh_off = (unsigned)(uintptr_t)(void*)&Bwh[0];
#endif

    FragC acc[6];   // 0..2: i_r,i_z,i_n   3..5: h_r,h_z,h_n
    for (int t = 0; t < 6; t++)
        for (int r = 0; r < 8; r++) acc[t].f[r] = 0.0f;

    for (int k0 = 0; k0 < OUT_CH; k0 += 32) {
        __syncthreads();
#if HAVE_TDM
        if (wave == 0)        // 3 gate slices x 64 rows x 32 bf16
            tdm_load_bf16_tile(bwi_off, wih16 + (size_t)c0 * 1024 + k0,
                               GRU_BC, 3, 1ull << 20, 1024);
        else if (wave == 1)
            tdm_load_bf16_tile(bwh_off, whh16 + (size_t)c0 * 1024 + k0,
                               GRU_BC, 3, 1ull << 20, 1024);
#else
        for (int j = 0; j < 6; j++) {
            int chunk = tid + j * 256;
            int mat = chunk / 768, rem = chunk % 768;
            int row = rem >> 2, q = rem & 3;
            int gate = row >> 6, n = row & 63;
            const unsigned short* w = mat ? whh16 : wih16;
            const unsigned short* sp =
                w + (size_t)(gate * 1024 + c0 + n) * 1024 + k0 + q * 8;
            uint4 v = *(const uint4*)sp;
            unsigned short* db = mat ? Bwh : Bwi;
            unsigned int* d = (unsigned int*)&db[row * LDSP + q * 8];
            d[0] = v.x; d[1] = v.y; d[2] = v.z; d[3] = v.w;
        }
#endif
        // stage A tiles (agg + h), 32x32 each, f32 -> bf16 (overlaps DMA)
        {
            int r = tid >> 3;           // 0..31
            int q = (tid & 7) * 4;      // 4 cols
            size_t off = (size_t)(m0 + r) * OUT_CH + k0 + q;
            float4 fa = *(const float4*)(agg + off);
            float4 fh = *(const float4*)(h + off);
            unsigned int* da = (unsigned int*)&Aag[r * LDSP + q];
            da[0] = pack_bf16x2(fa.x, fa.y);
            da[1] = pack_bf16x2(fa.z, fa.w);
            unsigned int* dh = (unsigned int*)&Ahh[r * LDSP + q];
            dh[0] = pack_bf16x2(fh.x, fh.y);
            dh[1] = pack_bf16x2(fh.z, fh.w);
        }
#if HAVE_TDM
        if (wave < 2) __builtin_amdgcn_s_wait_tensorcnt(0);
#endif
        __syncthreads();

        FragAB aa, ah;
        {
            const unsigned short* ra = &Aag[(wm + ln) * LDSP];
            const unsigned short* rh = &Ahh[(wm + ln) * LDSP];
            #pragma unroll
            for (int i = 0; i < 8; i++) {
                int kb = 2 * i + 16 * lg;
                aa.u[i] = *(const unsigned int*)&ra[kb];
                ah.u[i] = *(const unsigned int*)&rh[kb];
            }
        }
        #pragma unroll
        for (int gate = 0; gate < 3; gate++) {
            FragAB bi, bh;
            const unsigned short* ri = &Bwi[(gate * 64 + wc + ln) * LDSP];
            const unsigned short* rh = &Bwh[(gate * 64 + wc + ln) * LDSP];
            #pragma unroll
            for (int i = 0; i < 8; i++) {
                bi.u[i] = *(const unsigned int*)&ri[2 * i + 16 * lg];
                bh.u[i] = *(const unsigned int*)&rh[2 * i + 16 * lg];
            }
            acc[gate].v = __builtin_amdgcn_wmma_f32_16x16x32_bf16(
                false, aa.v, false, bi.v, (short)0, acc[gate].v, false, false);
            acc[3 + gate].v = __builtin_amdgcn_wmma_f32_16x16x32_bf16(
                false, ah.v, false, bh.v, (short)0, acc[3 + gate].v, false, false);
        }
    }

    // epilogue: gates
    int c = c0 + wc + ln;
    float bir = b_ih[c], biz = b_ih[1024 + c], bin = b_ih[2048 + c];
    float bhr = b_hh[c], bhz = b_hh[1024 + c], bhn = b_hh[2048 + c];
    #pragma unroll
    for (int r = 0; r < 8; r++) {
        int m = m0 + wm + r + 8 * lg;
        float hv = h[(size_t)m * OUT_CH + c];
        float rg = sigmoidf((acc[0].f[r] + bir) + (acc[3].f[r] + bhr));
        float zg = sigmoidf((acc[1].f[r] + biz) + (acc[4].f[r] + bhz));
        float ng = tanhf((acc[2].f[r] + bin) + rg * (acc[5].f[r] + bhn));
        hout[(size_t)m * OUT_CH + c] = (1.0f - zg) * ng + zg * hv;
    }
}

// ---------------------------------------------------------------------------
// 6) relu + per-graph per-channel online softmax aggregation.
//    batch is sorted: graph g owns nodes [g*256, (g+1)*256).
// ---------------------------------------------------------------------------
__global__ void softmax_agg_kernel(const float* __restrict__ h,
                                   const float* __restrict__ tptr,
                                   float* __restrict__ g) {
    int graph = blockIdx.x;
    int c = blockIdx.y * 256 + threadIdx.x;
    float t = tptr[0];
    size_t base = (size_t)graph * 256 * OUT_CH + c;
    float M = -3.4e38f, D = 0.0f, W = 0.0f;
    for (int nd = 0; nd < 256; nd++) {
        float hv = h[base + (size_t)nd * OUT_CH];
        hv = fmaxf(hv, 0.0f);                      // relu
        float s = t * hv;
        if (s > M) { float sc = __expf(M - s); D *= sc; W *= sc; M = s; }
        float e = __expf(s - M);
        D += e; W += e * hv;
    }
    g[(size_t)graph * OUT_CH + c] = W / D;
}

// ---------------------------------------------------------------------------
// 7) MLP head (tiny): one wave32 per output, shuffle reduction.
// ---------------------------------------------------------------------------
__global__ void head1_kernel(const float* __restrict__ g,
                             const float* __restrict__ w,
                             const float* __restrict__ b,
                             float* __restrict__ hid) {
    int out = blockIdx.x * 8 + (threadIdx.x >> 5);
    int lane = threadIdx.x & 31;
    int gr = out >> 9;        // / 512
    int j  = out & 511;
    const float* gv = g + (size_t)gr * OUT_CH;
    const float* wr = w + (size_t)j * OUT_CH;
    float s = 0.0f;
    for (int k = lane; k < OUT_CH; k += 32) s += gv[k] * wr[k];
    for (int off = 16; off > 0; off >>= 1) s += __shfl_down(s, off, 32);
    if (lane == 0) hid[out] = fmaxf(s + b[j], 0.0f);
}

__global__ void head2_kernel(const float* __restrict__ hid,
                             const float* __restrict__ ow,
                             const float* __restrict__ ob,
                             float* __restrict__ out) {
    int gr = blockIdx.x * 8 + (threadIdx.x >> 5);
    int lane = threadIdx.x & 31;
    const float* hv = hid + (size_t)gr * HID;
    float s = 0.0f;
    for (int k = lane; k < HID; k += 32) s += hv[k] * ow[k];
    for (int off = 16; off > 0; off >>= 1) s += __shfl_down(s, off, 32);
    if (lane == 0) out[gr] = s + ob[0];
}

// ---------------------------------------------------------------------------
extern "C" void kernel_launch(void* const* d_in, const int* in_sizes, int n_in,
                              void* d_out, int out_size, void* d_ws, size_t ws_size,
                              hipStream_t stream) {
    const float* x      = (const float*)d_in[0];
    const int*   ei     = (const int*)d_in[1];     // [2, N_EDGES]
    // d_in[2] = batch (unused: segments are contiguous 256-node blocks)
    const float* embed  = (const float*)d_in[3];
    const float* conv_w = (const float*)d_in[4];   // [2,1024,1024]
    const float* w_ih   = (const float*)d_in[5];   // [3072,1024]
    const float* w_hh   = (const float*)d_in[6];
    const float* b_ih   = (const float*)d_in[7];
    const float* b_hh   = (const float*)d_in[8];
    const float* tpar   = (const float*)d_in[9];
    const float* h1_w   = (const float*)d_in[10];  // [512,1024]
    const float* h1_b   = (const float*)d_in[11];
    const float* o_w    = (const float*)d_in[12];  // [1,512]
    const float* o_b    = (const float*)d_in[13];
    float* out = (float*)d_out;

    const int* e_src = ei;
    const int* e_dst = ei + N_EDGES;

    // workspace layout
    const size_t NH = (size_t)N_NODES * OUT_CH;
    float* hA  = (float*)d_ws;
    float* hB  = hA + NH;
    float* agg = hB + NH;
    unsigned short* cwT   = (unsigned short*)(agg + NH);      // [2][1024n][1024k]
    unsigned short* wih16 = cwT + (size_t)2 * 1024 * 1024;    // [3072][1024]
    unsigned short* whh16 = wih16 + (size_t)3072 * 1024;
    float* gbuf = (float*)(whh16 + (size_t)3072 * 1024);      // [256,1024]
    float* hid  = gbuf + (size_t)N_GRAPHS * OUT_CH;           // [256,512]

    // weights -> bf16 (conv_w transposed to n-major; w_ih/w_hh native layout)
    conv_w_transpose_bf16_kernel<<<(2 * 1024 * 1024) / 256, 256, 0, stream>>>(
        conv_w, cwT);
    to_bf16_kernel<<<(3072 * 1024) / 256, 256, 0, stream>>>(
        w_ih, wih16, 3072 * 1024);
    to_bf16_kernel<<<(3072 * 1024) / 256, 256, 0, stream>>>(
        w_hh, whh16, 3072 * 1024);

    // h init
    init_h_kernel<<<N_NODES, 256, 0, stream>>>(x, embed, hA);

    // 2 GatedGraphConv layers; ping-pong hA <-> hB.
    // Uses linearity: segment_sum((h@W)[src]) == segment_sum(h[src]) @ W
    float* cur = hA;
    float* nxt = hB;
    for (int l = 0; l < 2; l++) {
        (void)hipMemsetAsync(nxt, 0, NH * sizeof(float), stream);
        scatter_add_kernel<<<N_EDGES, 256, 0, stream>>>(cur, e_src, e_dst, nxt);
        // agg = scatter_sum @ conv_w[l]
        {
            dim3 grid(OUT_CH / GEMM_BN, N_NODES / GEMM_BM);
            gemm_bf16_wmma<<<grid, 256, 0, stream>>>(
                nxt, cwT + (size_t)l * 1024 * 1024, agg,
                N_NODES, OUT_CH, OUT_CH);
        }
        // h_next = GRU(agg, cur) -> nxt (scatter sums already consumed)
        {
            dim3 grid(OUT_CH / GRU_BC, N_NODES / GRU_BM);
            gru_fused_wmma<<<grid, 256, 0, stream>>>(
                agg, cur, wih16, whh16, b_ih, b_hh, nxt);
        }
        float* tmp = cur; cur = nxt; nxt = tmp;
    }
    // after 2 layers, final h is in `cur`

    // relu + softmax aggregation -> gbuf [256,1024]
    {
        dim3 grid(N_GRAPHS, OUT_CH / 256);
        softmax_agg_kernel<<<grid, 256, 0, stream>>>(cur, tpar, gbuf);
    }
    // MLP head
    head1_kernel<<<(N_GRAPHS * HID) / 8, 256, 0, stream>>>(gbuf, h1_w, h1_b, hid);
    head2_kernel<<<N_GRAPHS / 8, 256, 0, stream>>>(hid, o_w, o_b, out);
}
